// reblur_local_attention_67946382622891
// MI455X (gfx1250) — matrix operand order repaired and checked by reading, not compiled, not run on hardware
//
#include <hip/hip_runtime.h>
#include <hip/hip_bf16.h>
#include <math.h>

typedef __attribute__((ext_vector_type(16))) _Float16 v16h;
typedef __attribute__((ext_vector_type(8)))  float    v8f;

#define CIN   64
#define HH    256
#define WW    256
#define BB    4
#define OC    9          // output channels of every conv here (kk = 3*3)

// ---------------------------------------------------------------------------
// Pack conv weights [OC, Cin, KSZ, KSZ] (f32) into the exact per-lane A-matrix
// layout for V_WMMA_F32_16X16X32_F16, one 16-half fragment per lane, per
// (channel-chunk, spatial tap):  out[((chunk*KK+tap)*32 + lane)*16 + i]
// A layout used here (16-bit 16x32): lane m=lane%16, half-group hg=lane/16,
//   K = hg*16 + i (contiguous, matches B fragment)
// ---------------------------------------------------------------------------
__global__ void pack_w_kernel(const float* __restrict__ w, _Float16* __restrict__ out,
                              int KSZ, int Cin) {
    int KK = KSZ * KSZ;
    int total = 2 * KK * 32 * 16;               // 2 channel chunks of 32
    for (int idx = blockIdx.x * blockDim.x + threadIdx.x; idx < total;
         idx += gridDim.x * blockDim.x) {
        int i    = idx & 15;
        int t    = idx >> 4;
        int lane = t & 31;  t >>= 5;
        int tap  = t % KK;
        int chunk = t / KK;
        int m  = lane & 15;
        int hg = lane >> 4;
        int k  = hg * 16 + i;
        int ch = chunk * 32 + k;
        int dy = tap / KSZ, dx = tap % KSZ;
        float v = 0.0f;
        if (m < OC && ch < Cin)
            v = w[((m * Cin + ch) * KSZ + dy) * KSZ + dx];
        out[idx] = (_Float16)v;
    }
}

// ---------------------------------------------------------------------------
// Implicit-GEMM conv (Cin=64 -> OC=9, stride 1, pad KSZ/2) via WMMA f16.
// Block = 256 threads = 8 wave32; block covers 128 pixels of one image row.
// Each wave owns a 16-pixel N-tile; K is looped as 2 channel-chunks x KK taps.
// If src1 != nullptr the staged input is (src0 - src1)  (fuses d = x - att).
// ---------------------------------------------------------------------------
template <int KSZ, bool APPLY_TANH>
__global__ void conv_wmma_kernel(const float* __restrict__ src0,
                                 const float* __restrict__ src1,
                                 const _Float16* __restrict__ wpack,
                                 const float* __restrict__ bias,
                                 float* __restrict__ dst) {
    constexpr int PAD = KSZ / 2;
    constexpr int KK  = KSZ * KSZ;
    constexpr int TW  = 128 + KSZ - 1;     // staged columns
    constexpr int CH  = 32;                // channels per chunk
    constexpr int TOT = KSZ * TW * CH;

    __shared__ _Float16 lds[TOT] __attribute__((aligned(32)));

    const int tid  = threadIdx.x;
    const int wave = tid >> 5;
    const int lane = tid & 31;
    const int n    = lane & 15;            // N (pixel) index inside tile
    const int hi   = lane >> 4;            // K half-group for B fragment
    const int w0   = blockIdx.x * 128;     // pixel-column base of this block
    const int h    = blockIdx.y;
    const int b    = blockIdx.z;

    v8f acc = {};

    for (int chunk = 0; chunk < 2; ++chunk) {
        // ---- stage 32 channels x KSZ rows x TW cols into LDS (f16, ch inner)
        __syncthreads();
        const int cc = chunk * CH;
        for (int e = tid; e < TOT; e += 256) {
            int c   = e / (KSZ * TW);
            int rem = e - c * (KSZ * TW);
            int y   = rem / TW;
            int x   = rem - y * TW;
            int gy  = h + y - PAD;
            int gx  = w0 + x - PAD;
            float v = 0.0f;
            if (gy >= 0 && gy < HH && gx >= 0 && gx < WW) {
                int idx = ((b * CIN + cc + c) * HH + gy) * WW + gx;
                v = src0[idx];
                if (src1) v -= src1[idx];
            }
            lds[(y * TW + x) * CH + c] = (_Float16)v;
        }
        __syncthreads();

        // ---- K loop: one WMMA (K=32 channels) per spatial tap
        const _Float16* wp = wpack + (size_t)chunk * KK * 32 * 16;
#pragma unroll
        for (int tap = 0; tap < KK; ++tap) {
            v16h a = *(const v16h*)(wp + (tap * 32 + lane) * 16);
            int dy  = tap / KSZ;
            int dx  = tap - dy * KSZ;
            int col = wave * 16 + n + dx;                    // < TW by design
            v16h bf = *(const v16h*)&lds[(dy * TW + col) * CH + hi * 16];
            acc = __builtin_amdgcn_wmma_f32_16x16x32_f16(
                false, a, false, bf, (short)0, acc, false, false);
        }
    }

    // ---- epilogue: C/D layout VGPR r -> M = r + 8*hi, N = lane%16
    const int pw = w0 + wave * 16 + n;
#pragma unroll
    for (int r = 0; r < 8; ++r) {
        int m = r + hi * 8;
        if (m < OC) {
            float v = acc[r] + bias[m];
            if (APPLY_TANH) v = tanhf(v);
            dst[((b * OC + m) * HH + h) * WW + pw] = v;
        }
    }
}

// ---------------------------------------------------------------------------
// Per-pixel dynamic 3x3 filter: out[b,c,h,w] = sum_j x_pad[...]*katt[b,j,h,w]
// Pure bandwidth (katt reused 64x across c, taps hit L2 -> ~HBM roofline).
// ---------------------------------------------------------------------------
__global__ void per_pix_kernel(const float* __restrict__ feat,
                               const float* __restrict__ katt,
                               float* __restrict__ att_out) {
    int idx = blockIdx.x * 256 + threadIdx.x;       // B*C*H*W = 16,777,216
    int w = idx & 255;
    int t = idx >> 8;
    int hh = t & 255;  t >>= 8;
    int c = t & 63;
    int b = t >> 6;
    float s = 0.0f;
#pragma unroll
    for (int j = 0; j < 9; ++j) {
        int y = hh + j / 3 - 1;
        int x = w + j % 3 - 1;
        if ((unsigned)y < (unsigned)HH && (unsigned)x < (unsigned)WW)
            s += feat[((b * CIN + c) * HH + y) * WW + x] *
                 katt[((b * OC + j) * HH + hh) * WW + w];
    }
    att_out[idx] = s;
}

// ---------------------------------------------------------------------------
// Tiny 9->9 conv3x3 + tanh (81 MACs/out, memory-bound) -> final reblur filter
// ---------------------------------------------------------------------------
__global__ void conv99_tanh_kernel(const float* __restrict__ r1,
                                   const float* __restrict__ w,
                                   const float* __restrict__ bias,
                                   float* __restrict__ out) {
    int idx = blockIdx.x * 256 + threadIdx.x;       // B*9*H*W = 2,359,296
    int ww = idx & 255;
    int t = idx >> 8;
    int hh = t & 255;  t >>= 8;
    int m = t % OC;
    int b = t / OC;
    float s = bias[m];
#pragma unroll
    for (int c = 0; c < OC; ++c)
#pragma unroll
        for (int dy = 0; dy < 3; ++dy) {
            int y = hh + dy - 1;
            if ((unsigned)y >= (unsigned)HH) continue;
#pragma unroll
            for (int dx = 0; dx < 3; ++dx) {
                int x = ww + dx - 1;
                if ((unsigned)x >= (unsigned)WW) continue;
                s += w[((m * OC + c) * 3 + dy) * 3 + dx] *
                     r1[((b * OC + c) * HH + y) * WW + x];
            }
        }
    out[idx] = tanhf(s);
}

// ---------------------------------------------------------------------------
extern "C" void kernel_launch(void* const* d_in, const int* in_sizes, int n_in,
                              void* d_out, int out_size, void* d_ws, size_t ws_size,
                              hipStream_t stream) {
    const float* feature = (const float*)d_in[0];   // [4,64,256,256]
    const float* w_sharp = (const float*)d_in[1];   // [9,64,5,5]
    const float* b_sharp = (const float*)d_in[2];   // [9]
    const float* w_r1    = (const float*)d_in[3];   // [9,64,3,3]
    const float* b_r1    = (const float*)d_in[4];   // [9]
    const float* w_r2    = (const float*)d_in[5];   // [9,9,3,3]
    const float* b_r2    = (const float*)d_in[6];   // [9]

    float* out        = (float*)d_out;
    const int RB_N    = BB * OC * HH * WW;          // 2,359,296
    float* reblur_out = out;                        // output #1
    float* att_out    = out + RB_N;                 // output #2 [4,64,256,256]

    // workspace carve-up (~19 MB)
    char* ws = (char*)d_ws;
    _Float16* wp5   = (_Float16*)ws;                         // 2*25*512*2 = 51200 B
    _Float16* wp3   = (_Float16*)(ws + 51200);               // 2*9*512*2  = 18432 B
    float*    att_k = (float*)(ws + 69632);                  // [4,9,256,256] 9.4 MB
    float*    r1buf = (float*)(ws + 69632 + (size_t)RB_N * 4);

    // 1) pack both conv weight sets into WMMA A-fragment layout (f16)
    pack_w_kernel<<<dim3(64), dim3(256), 0, stream>>>(w_sharp, wp5, 5, CIN);
    pack_w_kernel<<<dim3(64), dim3(256), 0, stream>>>(w_r1,    wp3, 3, CIN);

    // 2) conv5x5(feature) + tanh -> attention kernel maps   [WMMA, 7.6 GF]
    conv_wmma_kernel<5, true><<<dim3(2, HH, BB), dim3(256), 0, stream>>>(
        feature, nullptr, wp5, b_sharp, att_k);

    // 3) per-pixel dynamic filter -> attention_feature (output #2)
    per_pix_kernel<<<dim3((BB * CIN * HH * WW) / 256), dim3(256), 0, stream>>>(
        feature, att_k, att_out);

    // 4) conv3x3(feature - attention_feature) -> r1         [WMMA, 2.7 GF]
    //    (subtraction fused into LDS staging; no 67 MB diff buffer)
    conv_wmma_kernel<3, false><<<dim3(2, HH, BB), dim3(256), 0, stream>>>(
        feature, att_out, wp3, b_r1, r1buf);

    // 5) conv3x3 9->9 + tanh -> reblur_filter (output #1)
    conv99_tanh_kernel<<<dim3(RB_N / 256), dim3(256), 0, stream>>>(
        r1buf, w_r2, b_r2, reblur_out);
}
